// MoEResNetBKLayer_66477503807728
// MI455X (gfx1250) — compile-verified, matrix operand order checked by reading, not run on hardware
//
#include <hip/hip_runtime.h>
#include <hip/hip_bf16.h>
#include <math.h>

#define D_MODEL 1024
#define N_SEQ   2048
#define BATCHSZ 2
#define NTOK    (BATCHSZ * N_SEQ)   // 4096 tokens
#define N_EXP   4
#define D_FF    4096
#define V_MAXC  3.0f
#define F_CLAMP 10.0f

#define MTILE   32                   // tokens per workgroup (2 WMMA M-tiles)
#define XSTR    1040                 // 1024 + 16 bf16 pad (keeps 16B align)
#define FCHUNK  1024                 // F processed per LDS round-trip
#define HSTR    (FCHUNK + 16)        // 1040
#define NCHUNK  (D_FF / FCHUNK)      // 4

typedef __bf16 bf16;
typedef __attribute__((ext_vector_type(8)))  bf16  bf16x8;
typedef __attribute__((ext_vector_type(16))) bf16  v16bf;
typedef __attribute__((ext_vector_type(8)))  float v8f;

#if defined(__gfx1250__) && __has_builtin(__builtin_amdgcn_global_load_async_to_lds_b128)
#define HAVE_ASYNC_COPY 1
typedef int v4i_t __attribute__((vector_size(16)));
typedef __attribute__((address_space(1))) v4i_t* g_v4i_ptr;
typedef __attribute__((address_space(3))) v4i_t* l_v4i_ptr;
#endif

// ---------------------------------------------------------------- helpers
// One lane's A/B fragment for v_wmma_*_16x16x32_bf16: row `ln15` of a tile,
// K elements {kb..kb+7} and {kb+16..kb+23} => two contiguous 16B chunks.
__device__ __forceinline__ v16bf frag_load(const bf16* rowbase, int off) {
  union { v16bf v; bf16x8 h[2]; } u;
  u.h[0] = *(const bf16x8*)(rowbase + off);
  u.h[1] = *(const bf16x8*)(rowbase + off + 16);
  return u.v;
}

__device__ __forceinline__ float fast_tanh(float x) {
#if __has_builtin(__builtin_amdgcn_tanhf)
  return __builtin_amdgcn_tanhf(x);     // native v_tanh_f32 on gfx1250
#else
  return tanhf(x);
#endif
}

__device__ __forceinline__ float gelu_tanh(float x) {
  const float k0 = 0.7978845608028654f;   // sqrt(2/pi)
  const float k1 = 0.044715f;
  float t = k0 * (x + k1 * x * x * x);
  return 0.5f * x * (1.0f + fast_tanh(t));
}

// 16B global->LDS copy: async (ASYNCcnt, no VGPR round-trip) when available.
__device__ __forceinline__ void copy16_g2l(const void* g, void* l) {
#ifdef HAVE_ASYNC_COPY
  __builtin_amdgcn_global_load_async_to_lds_b128(
      reinterpret_cast<g_v4i_ptr>(reinterpret_cast<uintptr_t>(g)),
      reinterpret_cast<l_v4i_ptr>(reinterpret_cast<uintptr_t>(l)), 0, 0);
#else
  *(uint4*)l = *(const uint4*)g;
#endif
}

__device__ __forceinline__ void copy_wait() {
#ifdef HAVE_ASYNC_COPY
#if __has_builtin(__builtin_amdgcn_s_wait_asynccnt)
  __builtin_amdgcn_s_wait_asynccnt(0);
#else
  asm volatile("s_wait_asynccnt 0x0" ::: "memory");
#endif
#endif
}

// ---------------------------------------------------------------- kernels
// fp32 source is read ONCE: non-temporal loads keep the 134 MB fp32 weights
// from evicting the 67 MB bf16 working set we want resident in L2.
__global__ void f32_to_bf16_kernel(const float* __restrict__ src,
                                   bf16* __restrict__ dst, int n) {
  int i = blockIdx.x * blockDim.x + threadIdx.x;
  int stride = gridDim.x * blockDim.x;
  for (; i < n; i += stride) dst[i] = (bf16)__builtin_nontemporal_load(src + i);
}

__global__ void init_counts_kernel(int* __restrict__ counts) {
  if (threadIdx.x < N_EXP) counts[threadIdx.x] = 0;
}

// One wave (32 lanes) per token: v-projection, gate logits, argmax route.
__global__ void gate_route_kernel(const float* __restrict__ x,
                                  const float* __restrict__ v_w,
                                  const float* __restrict__ v_b,
                                  const float* __restrict__ gate_w,
                                  const float* __restrict__ gate_b,
                                  float* __restrict__ he,
                                  int* __restrict__ counts,
                                  int* __restrict__ bucket) {
  int wave = threadIdx.x >> 5;
  int lane = threadIdx.x & 31;
  int token = blockIdx.x * 8 + wave;
  if (token >= NTOK) return;
  const float* xr = x + (size_t)token * D_MODEL;
  float sv = 0.f, s0 = 0.f, s1 = 0.f, s2 = 0.f, s3 = 0.f;
  for (int i = lane; i < D_MODEL; i += 32) {
    float xv = xr[i];
    sv += xv * v_w[i];
    s0 += xv * gate_w[i];
    s1 += xv * gate_w[D_MODEL + i];
    s2 += xv * gate_w[2 * D_MODEL + i];
    s3 += xv * gate_w[3 * D_MODEL + i];
  }
  for (int off = 16; off > 0; off >>= 1) {
    sv += __shfl_xor(sv, off, 32);
    s0 += __shfl_xor(s0, off, 32);
    s1 += __shfl_xor(s1, off, 32);
    s2 += __shfl_xor(s2, off, 32);
    s3 += __shfl_xor(s3, off, 32);
  }
  if (lane == 0) {
    float v = sv + v_b[0];
    v = fminf(fmaxf(v, -V_MAXC), V_MAXC);
    he[token] = v - 2.0f;
    float lg0 = s0 + gate_b[0], lg1 = s1 + gate_b[1];
    float lg2 = s2 + gate_b[2], lg3 = s3 + gate_b[3];
    int e = 0; float best = lg0;                 // strict '>' => first max wins
    if (lg1 > best) { best = lg1; e = 1; }
    if (lg2 > best) { best = lg2; e = 2; }
    if (lg3 > best) { best = lg3; e = 3; }
    int pos = atomicAdd(&counts[e], 1);
    bucket[e * NTOK + pos] = token;              // top-1 => routing weight = 1
  }
}

// Tridiagonal Green-function diagonal: forward + backward continued fractions
// (serial over N per batch row), then elementwise feature computation.
__global__ void bk_scan_kernel(const float* __restrict__ he,
                               const float* __restrict__ epsp,
                               const float* __restrict__ gammap,
                               float2* __restrict__ L,
                               float2* __restrict__ R,
                               float* __restrict__ feat) {
  float p  = epsp[0];
  float sp = (p > 20.f) ? p : log1pf(expf(p));   // softplus
  float zi = sp + 1e-6f + gammap[0];             // z = i*zi ; d = he - i*zi
  int tid = threadIdx.x;
  if (tid < 4) {
    int b = tid >> 1, back = tid & 1;
    const float* dr = he + b * N_SEQ;
    float cr = 1.f, ci = 0.f;                    // carry
    for (int k = 0; k < N_SEQ; ++k) {
      int n = back ? (N_SEQ - 1 - k) : k;
      float dre = dr[n], dim = -zi;
      float lr, li;
      if (k == 0) { lr = dre; li = dim; }        // c_0 = 0
      else {                                     // L = d - 1/carry
        float den = cr * cr + ci * ci;
        lr = dre - cr / den;
        li = dim + ci / den;
      }
      float2 val; val.x = lr; val.y = li;
      if (back) R[b * N_SEQ + n] = val; else L[b * N_SEQ + n] = val;
      cr = lr; ci = li;
    }
  }
  __syncthreads();
  for (int idx = tid; idx < NTOK; idx += blockDim.x) {
    float2 l = L[idx], r = R[idx];
    float dre = he[idx], dim = -zi;
    float ar = l.x + r.x - dre;
    float ai = l.y + r.y - dim;
    float den = ar * ar + ai * ai;
    float gr = ar / den, gi = -ai / den;         // G = 1/(L+R-d)
    gr = fminf(fmaxf(gr, -F_CLAMP), F_CLAMP);
    gi = fminf(fmaxf(gi, -F_CLAMP), F_CLAMP);
    feat[idx * 2]     = gr;
    feat[idx * 2 + 1] = gi;
  }
}

// Fused per-expert FFN: grid=(tiles, expert). Each workgroup = 32 tokens of
// one expert (two 16-row WMMA M-tiles sharing every B fragment); the h chunk
// (32 x FCHUNK bf16) stays in LDS between the two WMMA GEMMs.
__global__ __launch_bounds__(256)
void moe_ffn_wmma_kernel(const bf16* __restrict__ xb,
                         const bf16* __restrict__ w1b,
                         const float* __restrict__ b1,
                         const bf16* __restrict__ w2b,
                         const float* __restrict__ b2,
                         const float* __restrict__ out_w,
                         const float* __restrict__ out_b,
                         const float* __restrict__ bk_scale,
                         const float* __restrict__ feat,
                         const int* __restrict__ counts,
                         const int* __restrict__ bucket,
                         float* __restrict__ out) {
  extern __shared__ char smem[];
  bf16*  xlds    = (bf16*)smem;                                        // MTILE*XSTR
  bf16*  hlds    = (bf16*)(smem + MTILE * XSTR * 2);                   // MTILE*HSTR
  int*   tokLds  = (int*)(smem + MTILE * XSTR * 2 + MTILE * HSTR * 2); // MTILE
  float* featLds = (float*)(tokLds + MTILE);                           // MTILE*2

  int e = blockIdx.y;
  int cnt = counts[e];
  int tile0 = blockIdx.x * MTILE;
  if (tile0 >= cnt) return;                      // block-uniform: EXEC stays full

  int tid = threadIdx.x;
  if (tid < MTILE) {
    int g = tile0 + tid;
    int token = (g < cnt) ? bucket[e * NTOK + g] : -1;
    tokLds[tid] = token;
    int t0 = (token < 0) ? 0 : token;
    featLds[tid * 2]     = feat[t0 * 2];
    featLds[tid * 2 + 1] = feat[t0 * 2 + 1];
  }
  __syncthreads();

  { // gather MTILE token rows of x (bf16) into LDS, async 128-bit chunks
    int r   = tid >> 3;                           // 0..31 row
    int seg = tid & 7;                            // 8 threads per row
    int token = tokLds[r]; if (token < 0) token = 0;
    const uint4* src = (const uint4*)(xb + (size_t)token * D_MODEL);
    uint4* dst = (uint4*)(xlds + r * XSTR);
#pragma unroll
    for (int i = 0; i < 16; ++i) copy16_g2l(src + seg * 16 + i, dst + seg * 16 + i);
    copy_wait();
  }
  __syncthreads();

  int wave = tid >> 5;
  int lane = tid & 31;
  int ln15 = lane & 15;
  int kb   = (lane & 16) ? 8 : 0;                 // K-half per ISA 16-bit layout
  int msel = (lane & 16) ? 8 : 0;                 // C/D row offset per lane-half

  // persistent output accumulators: this wave owns D columns [dbase, dbase+128)
  // for both 16-row M-tiles.
  int dbase = wave * 128;
  v8f yacc[2][8];
#pragma unroll
  for (int j = 0; j < 8; ++j) {
    float bb = b2[e * D_MODEL + dbase + j * 16 + ln15];
#pragma unroll
    for (int r = 0; r < 8; ++r) { yacc[0][j][r] = bb; yacc[1][j][r] = bb; }
  }

  // hoisted W2 row bases for this wave's 8 D sub-tiles
  const bf16* w2base[8];
#pragma unroll
  for (int j = 0; j < 8; ++j)
    w2base[j] = w2b + ((size_t)e * D_MODEL + dbase + j * 16 + ln15) * D_FF;

  const bf16* xrow0 = xlds + ln15 * XSTR;
  const bf16* xrow1 = xlds + (16 + ln15) * XSTR;
  const bf16* hrow0 = hlds + ln15 * HSTR;
  const bf16* hrow1 = hlds + (16 + ln15) * HSTR;

  for (int c = 0; c < NCHUNK; ++c) {
    // ---- phase 1: h[:, c*FCHUNK + wave*128 .. +128) = gelu(x @ W1^T + b1)
#pragma unroll 1
    for (int s = 0; s < 8; ++s) {
      int fLocal = wave * 128 + s * 16;
      int fGlob  = c * FCHUNK + fLocal;
      float bias = b1[e * D_FF + fGlob + ln15];
      v8f acc0, acc1;
#pragma unroll
      for (int r = 0; r < 8; ++r) { acc0[r] = bias; acc1[r] = bias; }
      const bf16* w1row = w1b + ((size_t)e * D_FF + fGlob + ln15) * D_MODEL;
#pragma unroll 4
      for (int d0 = 0; d0 < D_MODEL; d0 += 32) {
        v16bf bfrag = frag_load(w1row, d0 + kb);      // shared by both M-tiles
        v16bf a0 = frag_load(xrow0, d0 + kb);
        v16bf a1 = frag_load(xrow1, d0 + kb);
        acc0 = __builtin_amdgcn_wmma_f32_16x16x32_bf16(
            false, a0, false, bfrag, (short)0, acc0, false, false);
        acc1 = __builtin_amdgcn_wmma_f32_16x16x32_bf16(
            false, a1, false, bfrag, (short)0, acc1, false, false);
      }
#pragma unroll
      for (int r = 0; r < 8; ++r) {
        hlds[(r + msel) * HSTR + fLocal + ln15]        = (bf16)gelu_tanh(acc0[r]);
        hlds[(16 + r + msel) * HSTR + fLocal + ln15]   = (bf16)gelu_tanh(acc1[r]);
      }
    }
    __syncthreads();

    // ---- phase 2: y[:, dbase..dbase+128) += h_chunk @ W2_chunk^T
#pragma unroll 1
    for (int f0 = 0; f0 < FCHUNK; f0 += 32) {
      v16bf a0 = frag_load(hrow0, f0 + kb);
      v16bf a1 = frag_load(hrow1, f0 + kb);
#pragma unroll
      for (int j = 0; j < 8; ++j) {
        v16bf bfrag = frag_load(w2base[j] + c * FCHUNK, f0 + kb);
        yacc[0][j] = __builtin_amdgcn_wmma_f32_16x16x32_bf16(
            false, a0, false, bfrag, (short)0, yacc[0][j], false, false);
        yacc[1][j] = __builtin_amdgcn_wmma_f32_16x16x32_bf16(
            false, a1, false, bfrag, (short)0, yacc[1][j], false, false);
      }
    }
    __syncthreads();
  }

  // ---- epilogue: + bk_scale * (feat @ out_w^T + out_b), scatter to tokens.
  // Output is write-once / never re-read: non-temporal stores spare L2.
#pragma unroll
  for (int j = 0; j < 8; ++j) {
    int d = dbase + j * 16 + ln15;
    float ow0 = out_w[d * 2], ow1 = out_w[d * 2 + 1];
    float ob  = out_b[d],     bks = bk_scale[d];
#pragma unroll
    for (int mt = 0; mt < 2; ++mt) {
#pragma unroll
      for (int r = 0; r < 8; ++r) {
        int m = mt * 16 + r + msel;
        int token = tokLds[m];
        if (token >= 0) {
          float spec = featLds[m * 2] * ow0 + featLds[m * 2 + 1] * ow1 + ob;
          __builtin_nontemporal_store(yacc[mt][j][r] + bks * spec,
                                      out + (size_t)token * D_MODEL + d);
        }
      }
    }
  }
}

// ---------------------------------------------------------------- launcher
extern "C" void kernel_launch(void* const* d_in, const int* in_sizes, int n_in,
                              void* d_out, int out_size, void* d_ws, size_t ws_size,
                              hipStream_t stream) {
  const float* x        = (const float*)d_in[0];
  const float* v_w      = (const float*)d_in[1];
  const float* v_b      = (const float*)d_in[2];
  const float* gate_w   = (const float*)d_in[3];
  const float* gate_b   = (const float*)d_in[4];
  const float* w1       = (const float*)d_in[5];
  const float* b1       = (const float*)d_in[6];
  const float* w2       = (const float*)d_in[7];
  const float* b2       = (const float*)d_in[8];
  const float* out_w    = (const float*)d_in[9];
  const float* out_b    = (const float*)d_in[10];
  const float* bk_scale = (const float*)d_in[11];
  const float* epsp     = (const float*)d_in[12];
  const float* gammap   = (const float*)d_in[13];
  float* out = (float*)d_out;

  // workspace layout
  char* ws = (char*)d_ws;
  size_t off = 0;
  bf16* xb   = (bf16*)(ws + off); off += (size_t)NTOK * D_MODEL * 2;
  bf16* w1b  = (bf16*)(ws + off); off += (size_t)N_EXP * D_FF * D_MODEL * 2;
  bf16* w2b  = (bf16*)(ws + off); off += (size_t)N_EXP * D_MODEL * D_FF * 2;
  float* he  = (float*)(ws + off); off += (size_t)NTOK * 4;
  float* feat= (float*)(ws + off); off += (size_t)NTOK * 2 * 4;
  float2* Lw = (float2*)(ws + off); off += (size_t)NTOK * 8;
  float2* Rw = (float2*)(ws + off); off += (size_t)NTOK * 8;
  int* counts= (int*)(ws + off);   off += 256;
  int* bucket= (int*)(ws + off);   off += (size_t)N_EXP * NTOK * 4;

  // 1) fp32 -> bf16 staging (bf16 weights land in L2; 67 MB fits in 192 MB)
  f32_to_bf16_kernel<<<1024, 256, 0, stream>>>(x,  xb,  NTOK * D_MODEL);
  f32_to_bf16_kernel<<<2048, 256, 0, stream>>>(w1, w1b, N_EXP * D_FF * D_MODEL);
  f32_to_bf16_kernel<<<2048, 256, 0, stream>>>(w2, w2b, N_EXP * D_MODEL * D_FF);

  // 2) routing + v-projection
  init_counts_kernel<<<1, 32, 0, stream>>>(counts);
  gate_route_kernel<<<NTOK / 8, 256, 0, stream>>>(x, v_w, v_b, gate_w, gate_b,
                                                  he, counts, bucket);

  // 3) tridiagonal Green-function scan (tiny, serial over N)
  bk_scan_kernel<<<1, 64, 0, stream>>>(he, epsp, gammap, Lw, Rw, feat);

  // 4) fused top-1 MoE FFN with WMMA + spectral epilogue
  size_t smem = (size_t)MTILE * XSTR * 2 + (size_t)MTILE * HSTR * 2
              + MTILE * 4 + MTILE * 2 * 4;
  dim3 grid(NTOK / MTILE, N_EXP);
  moe_ffn_wmma_kernel<<<grid, 256, smem, stream>>>(xb, w1b, b1, w2b, b2,
                                                   out_w, out_b, bk_scale, feat,
                                                   counts, bucket, out);
}
